// OriginalGudi_UpProj_Block_Cat_13580686590345
// MI455X (gfx1250) — compile-verified, matrix operand order checked by reading, not run on hardware
//
#include <hip/hip_runtime.h>
#include <hip/hip_bf16.h>

// ---------------- types ----------------
typedef __attribute__((ext_vector_type(16))) __bf16 v16bf;
typedef __attribute__((ext_vector_type(8)))  float  v8f;
typedef __attribute__((ext_vector_type(4)))  int    v4i;

struct U4 { unsigned int x, y, z, w; };           // trivial 16-byte chunk (union-safe)

#define HH   120
#define WW   160
#define HWP  (HH * WW)          // 19200
#define BB   8
#define NCOL (BB * HWP)         // 153600 output columns (B*H*W)
#define CNTF 153600.0f          // BN count: N*H*W
#define EPSV 1e-5f

// bf16 round-to-nearest-even (manual; avoids header API variance)
__device__ __forceinline__ unsigned short f2bf(float x) {
    union { float f; unsigned int u; } c; c.f = x;
    unsigned int r = c.u + 0x7FFFu + ((c.u >> 16) & 1u);
    return (unsigned short)(r >> 16);
}

// ---- async global->LDS staging helpers (CDNA5), with sync fallback ----
#if defined(__has_builtin) && __has_builtin(__builtin_amdgcn_global_load_async_to_lds_b128)
#define HAVE_ASYNC_LDS 1
#else
#define HAVE_ASYNC_LDS 0
#endif

__device__ __forceinline__ void lds_copy16(void* lds, const void* g) {
#if HAVE_ASYNC_LDS
    // this toolchain declares the builtin with generic int4* parameters
    __builtin_amdgcn_global_load_async_to_lds_b128((v4i*)g, (v4i*)lds, 0, 0);
#else
    *(U4*)lds = *(const U4*)g;
#endif
}
__device__ __forceinline__ void wait_async(int pending) {
#if HAVE_ASYNC_LDS
#if defined(__has_builtin) && __has_builtin(__builtin_amdgcn_s_wait_asynccnt)
    if (pending) __builtin_amdgcn_s_wait_asynccnt(2);
    else         __builtin_amdgcn_s_wait_asynccnt(0);
#else
    if (pending) { asm volatile("s_wait_asynccnt 2" ::: "memory"); }
    else         { asm volatile("s_wait_asynccnt 0" ::: "memory"); }
#endif
#else
    (void)pending;
#endif
}

// ---------------- small utility kernels ----------------
__global__ void zero_kernel(float* p, int n) {
    int i = blockIdx.x * blockDim.x + threadIdx.x;
    if (i < n) p[i] = 0.0f;
}

__global__ void cvt_w_kernel(const float* __restrict__ src,
                             unsigned short* __restrict__ dst, int n) {
    int i = blockIdx.x * blockDim.x + threadIdx.x;
    if (i < n) dst[i] = f2bf(src[i]);
}

// x [8,256,60,80] -> xu bf16 [8,256,120,160], checkerboard (even h,w positions)
__global__ void expand_xu_kernel(const float* __restrict__ x,
                                 unsigned short* __restrict__ xu) {
    long i = (long)blockIdx.x * 256 + threadIdx.x;   // over 8*256*19200
    int hw = (int)(i % HWP);
    long bc = i / HWP;
    int oh = hw / WW, ow = hw % WW;
    unsigned short v = 0;
    if (!(oh & 1) && !(ow & 1))
        v = f2bf(x[bc * (60 * 80) + (oh >> 1) * 80 + (ow >> 1)]);
    xu[i] = v;
}

// ---------------- implicit-GEMM conv with WMMA bf16 ----------------
// in : bf16 [B][CIN][HW]   wt : bf16 [CoutT][CIN*KH*KW]   out : f32 [B][CoutT][HW]
// WG tile: 128 (M, Cout) x 128 (N, spatial*batch); 8 waves, each 16M x 128N.
// Weights double-buffered via async global->LDS; im2col gathered synchronously.
template <int CIN, int KH, int KW, int PAD>
__launch_bounds__(256)
__global__ void conv_wmma_kernel(const unsigned short* __restrict__ in,
                                 const unsigned short* __restrict__ wt,
                                 float* __restrict__ out,
                                 float* __restrict__ sum,
                                 float* __restrict__ sumsq,
                                 int CoutT) {
    constexpr int KHW  = KH * KW;
    constexpr int Ktot = CIN * KHW;                 // multiple of 32 for all our convs

    __shared__ __align__(16) unsigned short As[2][128 * 32];  // weights [m][k], double buffered
    __shared__ __align__(16) unsigned short Bs[128 * 32];     // im2col  [n][k]

    const int t    = threadIdx.x;
    const int lane = t & 31;
    const int wid  = t >> 5;          // 0..7 (8 waves)
    const int half = lane >> 4;       // 0/1
    const int ln16 = lane & 15;

    const int colBase = blockIdx.x * 128;           // N tile (spatial*batch)
    const int mBase   = blockIdx.y * 128;           // M supertile (Cout)

    // fixed per-thread im2col coordinates (n fixed, kk varies with k0)
    const int nB  = t >> 1;                         // 0..127
    const int kkB = (t & 1) * 16;                   // 0 / 16
    const int colB = colBase + nB;
    const int bB  = colB / HWP;
    const int hwB = colB % HWP;
    const int ohB = hwB / WW;
    const int owB = hwB % WW;

    // per-thread A-stage coordinates: row t>>1, two U4 segments at (t&1)*2
    const int rowA = t >> 1;
    const int segA = (t & 1) * 2;

    v8f acc[8] = {};

    U4*       Bs4 = (U4*)Bs;
    const U4* wt4 = (const U4*)wt;

    auto issueA = [&](int k0, int bfi) {
        long base = (((long)(mBase + rowA) * Ktot + k0) >> 3);   // U4 units
        U4* dst = ((U4*)As[bfi]) + rowA * 4 + segA;
        lds_copy16(dst + 0, wt4 + base + segA + 0);
        lds_copy16(dst + 1, wt4 + base + segA + 1);
    };

    issueA(0, 0);
    int bf = 0;

    for (int k0 = 0; k0 < Ktot; k0 += 32) {
        // ---- stage B (im2col) tile: 128 x 32 bf16 = 8 KB (sync; needs OOB zero-fill) ----
        #pragma unroll
        for (int j = 0; j < 16; ++j) {
            int k   = k0 + kkB + j;
            int c   = k / KHW;                // constant divisor -> mul/shift
            int rem = k - c * KHW;
            int kh  = rem / KW;
            int kw  = rem - kh * KW;
            int ih  = ohB - PAD + kh;
            int iw  = owB - PAD + kw;
            unsigned short v = 0;
            if ((unsigned)ih < (unsigned)HH && (unsigned)iw < (unsigned)WW)
                v = in[((long)bB * CIN + c) * HWP + ih * WW + iw];
            Bs[nB * 32 + kkB + j] = v;
        }

        const int has_next = (k0 + 32) < Ktot;
        if (has_next) issueA(k0 + 32, bf ^ 1);     // overlap next weight tile
        wait_async(has_next);                      // current A-tile landed in LDS
        __syncthreads();

        // ---- 8 WMMAs per wave: 16x16 tiles over N, shared A fragment ----
        union { U4 u[2]; v16bf v; } fa, fb;
        U4* As4 = (U4*)As[bf];
        const int mrow = wid * 16 + ln16;
        fa.u[0] = As4[mrow * 4 + half];       // K = 8*half .. +7
        fa.u[1] = As4[mrow * 4 + 2 + half];   // K = 16+8*half .. +7
        #pragma unroll
        for (int nt = 0; nt < 8; ++nt) {
            const int n = nt * 16 + ln16;
            fb.u[0] = Bs4[n * 4 + half];
            fb.u[1] = Bs4[n * 4 + 2 + half];
            acc[nt] = __builtin_amdgcn_wmma_f32_16x16x32_bf16(
                false, fa.v, false, fb.v, (short)0, acc[nt], false, false);
        }
        __syncthreads();
        bf ^= 1;
    }

    // ---- epilogue: store raw conv output + per-channel BN statistics ----
    float s[8], q[8];
    #pragma unroll
    for (int r = 0; r < 8; ++r) { s[r] = 0.f; q[r] = 0.f; }

    #pragma unroll
    for (int nt = 0; nt < 8; ++nt) {
        int col = colBase + nt * 16 + ln16;
        int b   = col / HWP;
        int hw  = col % HWP;
        #pragma unroll
        for (int r = 0; r < 8; ++r) {
            int m = mBase + wid * 16 + r + 8 * half;
            float v = acc[nt][r];
            out[((long)b * CoutT + m) * HWP + hw] = v;
            s[r] += v;
            q[r] += v * v;
        }
    }
    // lanes 0..15 (and 16..31) share the same 8 channels -> butterfly reduce
    #pragma unroll
    for (int r = 0; r < 8; ++r) {
        #pragma unroll
        for (int off = 1; off < 16; off <<= 1) {
            s[r] += __shfl_xor(s[r], off, 32);
            q[r] += __shfl_xor(q[r], off, 32);
        }
        if (ln16 == 0) {
            int m = mBase + wid * 16 + r + 8 * half;
            atomicAdd(&sum[m],   s[r]);
            atomicAdd(&sumsq[m], q[r]);
        }
    }
}

// ---------------- BN apply kernels ----------------
__device__ __forceinline__ float bn_apply(float x, float s, float sq, float g, float b) {
    float m   = s * (1.0f / CNTF);
    float var = sq * (1.0f / CNTF) - m * m;
    return (x - m) * rsqrtf(var + EPSV) * g + b;
}

// cat[b][ch][hw]: ch<128 -> relu(bn(conv1)), ch>=128 -> side_input   (bf16)
__global__ void bn_relu_cat_kernel(const float* __restrict__ conv,   // [B][256][HW] (ch<128 = conv1)
                                   const float* __restrict__ side,   // [B][128][HW]
                                   const float* __restrict__ sum, const float* __restrict__ sq,
                                   const float* __restrict__ g,   const float* __restrict__ b,
                                   unsigned short* __restrict__ cat) {
    long i = (long)blockIdx.x * 256 + threadIdx.x;   // over 8*256*19200
    int hw = (int)(i % HWP);
    int ch = (int)((i / HWP) % 256);
    int bb = (int)(i / ((long)HWP * 256));
    float v;
    if (ch < 128) {
        float x = conv[((long)bb * 256 + ch) * HWP + hw];
        v = fmaxf(bn_apply(x, sum[ch], sq[ch], g[ch], b[ch]), 0.0f);
    } else {
        v = side[((long)bb * 128 + (ch - 128)) * HWP + hw];
    }
    cat[i] = f2bf(v);
}

// dst bf16 = relu(bn(src))   over [B][128][HW]
__global__ void bn_relu_kernel(const float* __restrict__ src,
                               const float* __restrict__ sum, const float* __restrict__ sq,
                               const float* __restrict__ g,   const float* __restrict__ b,
                               unsigned short* __restrict__ dst) {
    long i = (long)blockIdx.x * 256 + threadIdx.x;   // over 8*128*19200
    int ch = (int)((i / HWP) % 128);
    float v = fmaxf(bn_apply(src[i], sum[ch], sq[ch], g[ch], b[ch]), 0.0f);
    dst[i] = f2bf(v);
}

// out = relu(bn2(conv2) + bnsc(conv_sc))
__global__ void final_kernel(const float* __restrict__ conv2,   // [B][128][HW]
                             const float* __restrict__ conv1sc, // [B][256][HW], ch>=128 = shortcut
                             const float* __restrict__ sum2, const float* __restrict__ sq2,
                             const float* __restrict__ g2,   const float* __restrict__ b2,
                             const float* __restrict__ sumsc, const float* __restrict__ sqsc,
                             const float* __restrict__ gsc,  const float* __restrict__ bsc,
                             float* __restrict__ out) {
    long i = (long)blockIdx.x * 256 + threadIdx.x;   // over 8*128*19200
    int hw = (int)(i % HWP);
    int ch = (int)((i / HWP) % 128);
    int bb = (int)(i / ((long)HWP * 128));
    float v2 = bn_apply(conv2[i], sum2[ch], sq2[ch], g2[ch], b2[ch]);
    float vs = bn_apply(conv1sc[((long)bb * 256 + 128 + ch) * HWP + hw],
                        sumsc[ch], sqsc[ch], gsc[ch], bsc[ch]);
    out[i] = fmaxf(v2 + vs, 0.0f);
}

// ---------------- host ----------------
extern "C" void kernel_launch(void* const* d_in, const int* in_sizes, int n_in,
                              void* d_out, int out_size, void* d_ws, size_t ws_size,
                              hipStream_t stream) {
    const float* x    = (const float*)d_in[0];
    const float* side = (const float*)d_in[1];
    const float* w1   = (const float*)d_in[2];
    const float* g1   = (const float*)d_in[3];
    const float* b1   = (const float*)d_in[4];
    const float* w11  = (const float*)d_in[5];
    const float* g11  = (const float*)d_in[6];
    const float* b11  = (const float*)d_in[7];
    const float* w2   = (const float*)d_in[8];
    const float* g2   = (const float*)d_in[9];
    const float* b2   = (const float*)d_in[10];
    const float* wsc  = (const float*)d_in[11];
    const float* gsc  = (const float*)d_in[12];
    const float* bsc  = (const float*)d_in[13];
    float* out = (float*)d_out;
    (void)in_sizes; (void)n_in; (void)out_size; (void)ws_size;

    char* wsb = (char*)d_ws;
    size_t off = 0;
    auto alloc = [&](size_t bytes) -> void* {
        void* p = wsb + off;
        off += (bytes + 255) & ~(size_t)255;
        return p;
    };
    float*          STATS = (float*)alloc(1024 * sizeof(float));
    unsigned short* WCAT  = (unsigned short*)alloc((size_t)256 * 6400 * 2); // w1 || wsc
    unsigned short* W11B  = (unsigned short*)alloc((size_t)128 * 2304 * 2);
    unsigned short* W2B   = (unsigned short*)alloc((size_t)128 * 1152 * 2);
    unsigned short* XU    = (unsigned short*)alloc((size_t)8 * 256 * HWP * 2);
    float*          C1SC  = (float*)alloc((size_t)8 * 256 * HWP * 4);
    unsigned short* CAT   = (unsigned short*)alloc((size_t)8 * 256 * HWP * 2);
    float*          C11   = (float*)alloc((size_t)8 * 128 * HWP * 4);
    unsigned short* C2IN  = (unsigned short*)alloc((size_t)8 * 128 * HWP * 2);
    float*          C2    = (float*)alloc((size_t)8 * 128 * HWP * 4);

    float* sum1sc = STATS;         float* sq1sc = STATS + 256;   // 256 ch (conv1 + shortcut)
    float* sum11  = STATS + 512;   float* sq11  = STATS + 640;
    float* sum2   = STATS + 768;   float* sq2   = STATS + 896;

    // stats must be zeroed every call (harness does not re-poison)
    zero_kernel<<<4, 256, 0, stream>>>(STATS, 1024);

    // weights -> bf16 (OIHW flatten == [Cout][K] row-major)
    cvt_w_kernel<<<(128 * 6400 + 255) / 256, 256, 0, stream>>>(w1,  WCAT,              128 * 6400);
    cvt_w_kernel<<<(128 * 6400 + 255) / 256, 256, 0, stream>>>(wsc, WCAT + 128 * 6400, 128 * 6400);
    cvt_w_kernel<<<(128 * 2304 + 255) / 256, 256, 0, stream>>>(w11, W11B,              128 * 2304);
    cvt_w_kernel<<<(128 * 1152 + 255) / 256, 256, 0, stream>>>(w2,  W2B,               128 * 1152);

    // checkerboard upsample to bf16
    expand_xu_kernel<<<(8 * 256 * HWP) / 256, 256, 0, stream>>>(x, XU);

    // conv1 + shortcut fused (shared im2col), Cout=256
    conv_wmma_kernel<256, 5, 5, 2><<<dim3(NCOL / 128, 2), 256, 0, stream>>>(
        XU, WCAT, C1SC, sum1sc, sq1sc, 256);

    // bn+relu on conv1, concat with side -> bf16
    bn_relu_cat_kernel<<<(8 * 256 * HWP) / 256, 256, 0, stream>>>(
        C1SC, side, sum1sc, sq1sc, g1, b1, CAT);

    // conv11 (Cin=256, 3x3)
    conv_wmma_kernel<256, 3, 3, 1><<<dim3(NCOL / 128, 1), 256, 0, stream>>>(
        CAT, W11B, C11, sum11, sq11, 128);

    // bn+relu -> bf16
    bn_relu_kernel<<<(8 * 128 * HWP) / 256, 256, 0, stream>>>(
        C11, sum11, sq11, g11, b11, C2IN);

    // conv2 (Cin=128, 3x3)
    conv_wmma_kernel<128, 3, 3, 1><<<dim3(NCOL / 128, 1), 256, 0, stream>>>(
        C2IN, W2B, C2, sum2, sq2, 128);

    // out = relu(bn2(conv2) + bnsc(shortcut))
    final_kernel<<<(8 * 128 * HWP) / 256, 256, 0, stream>>>(
        C2, C1SC, sum2, sq2, g2, b2, sum1sc + 128, sq1sc + 128, gsc, bsc, out);
}
